// EquiformerV2Conv_26585847562375
// MI455X (gfx1250) — matrix operand order repaired and checked by reading, not compiled, not run on hardware
//
#include <hip/hip_runtime.h>

// ---------------------------------------------------------------------------
// EquiformerV2 attention block for MI455X (gfx1250, wave32, WMMA bf16)
//
// Pipeline (all on `stream`):
//   0. rot kernel (independent elementwise, E edges)
//   1. zero qb/kb (padding of head dim 144->160 must be 0)
//   2. stage W_{q,k,v,o} f32 -> bf16
//   3. stage x -> xt   (per-l [N*d, C] row-major bf16, c contiguous)
//   4. GEMM<LI,MODE> Q/K/V: y = xt * W^T * (1/sqrt C) via wmma bf16
//        A tile staged to LDS via TDM tensor_load_to_lds (s_wait_tensorcnt)
//        Q,K written [b,h,m,160] bf16 ; V written transposed [b,h,160,m] bf16
//   5. attention: flash streaming softmax, one wave per 16-q tile,
//        S = Q K^T (WMMA), P through LDS (ds_store_b16 / s_wait_dscnt /
//        ds_load_b128) into A-fragments, O += P V (WMMA), writes ot per-l
//   6. GEMM<LI,3>: attn_out = ot * Wo^T / sqrt(C) + x  -> d_out (f32)
//   7. LayerNorm+SiLU on first 128 channels of each row (in place)
//
// Workspace layout (bytes), total 101,056,512:
//   xt  @ 0          size 18,874,368   (N*9*C bf16)
//   qb  @ 18874368   size 20,971,520   (B*H*M*160 bf16)
//   kb  @ 39845888   size 20,971,520
//   vt  @ 60817408   size 20,971,520
//   ot  @ 81788928   size 18,874,368
//   Wb  @ 100663296  size    393,216   ([4][3][128][128] bf16)
// ---------------------------------------------------------------------------

typedef __attribute__((ext_vector_type(16))) __bf16        v16bf;
typedef __attribute__((ext_vector_type(8)))  float         v8f;
typedef __attribute__((ext_vector_type(4)))  unsigned int  uv4;
typedef __attribute__((ext_vector_type(4)))  unsigned int  u32x4;
typedef __attribute__((ext_vector_type(8)))  int           i32x8;
typedef __attribute__((ext_vector_type(4)))  int           i32x4;

#define NN    8192
#define DD    1152
#define CC    128
#define HH    8
#define BB    16
#define MM    512
#define EE    262144
#define HD    144
#define HDP   160   // padded head dim (5 chunks of 32)

__device__ __forceinline__ unsigned short f2bf(float x) {
  unsigned int u = __float_as_uint(x);
  unsigned int r = (u + 0x7FFFu + ((u >> 16) & 1u)) >> 16;   // RNE
  return (unsigned short)r;
}

union Frag { uv4 u[2]; v16bf v; };

__device__ __forceinline__ v16bf load_frag2(const unsigned short* p0,
                                            const unsigned short* p1) {
  Frag t;
  t.u[0] = *reinterpret_cast<const uv4*>(p0);
  t.u[1] = *reinterpret_cast<const uv4*>(p1);
  return t.v;
}
__device__ __forceinline__ v16bf load_frag32B(const unsigned short* p) {
  return load_frag2(p, p + 8);
}
__device__ __forceinline__ v8f vzero8() {
  v8f z;
#pragma unroll
  for (int i = 0; i < 8; ++i) z[i] = 0.0f;
  return z;
}
__device__ __forceinline__ v8f wmma_bf16(v16bf a, v16bf b, v8f c) {
  return __builtin_amdgcn_wmma_f32_16x16x32_bf16(false, a, false, b,
                                                 (short)0, c, false, false);
}

// map feature index f (0..1151) -> element offset in per-l [N*d,C] layout
__device__ __forceinline__ size_t irrep_index(int n, int f) {
  if (f < 128) {
    return (size_t)n * 128 + f;                      // l=0, d=1
  } else if (f < 512) {
    int ff = f - 128;                                // l=1, d=3
    return (size_t)1048576 + ((size_t)n * 3 + (ff % 3)) * 128 + (ff / 3);
  } else {
    int ff = f - 512;                                // l=2, d=5
    return (size_t)4194304 + ((size_t)n * 5 + (ff % 5)) * 128 + (ff / 5);
  }
}

// ---------------------------------------------------------------------------
// 0. edge rotation matrices
// ---------------------------------------------------------------------------
__global__ __launch_bounds__(256) void rot_kernel(const float* __restrict__ ev,
                                                  const float* __restrict__ er,
                                                  float* __restrict__ out) {
  int e = blockIdx.x * 256 + threadIdx.x;
  if (e >= EE) return;
  float vx = ev[e * 3], vy = ev[e * 3 + 1], vz = ev[e * 3 + 2];
  float inv = rsqrtf(vx * vx + vy * vy + vz * vz);
  float nx0 = vx * inv, nx1 = vy * inv, nx2 = vz * inv;
  float rx = er[e * 3] - 0.5f, ry = er[e * 3 + 1] - 0.5f, rz = er[e * 3 + 2] - 0.5f;
  float ri = rsqrtf(rx * rx + ry * ry + rz * rz);
  float e20 = rx * ri, e21 = ry * ri, e22 = rz * ri;
  float b0 = -e21, b1 = e20, b2 = e22;
  float c0 = e20, c1 = -e22, c2 = e21;
  auto adot = [&](float a0, float a1, float a2) {
    return fabsf(a0 * nx0 + a1 * nx1 + a2 * nx2);
  };
  float vdb = adot(b0, b1, b2), vdc = adot(c0, c1, c2);
  if (adot(e20, e21, e22) > vdb) { e20 = b0; e21 = b1; e22 = b2; }
  if (adot(e20, e21, e22) > vdc) { e20 = c0; e21 = c1; e22 = c2; }
  float z0 = nx1 * e22 - nx2 * e21;
  float z1 = nx2 * e20 - nx0 * e22;
  float z2 = nx0 * e21 - nx1 * e20;
  float zi = rsqrtf(z0 * z0 + z1 * z1 + z2 * z2);
  z0 *= zi; z1 *= zi; z2 *= zi;
  float y0 = nx1 * z2 - nx2 * z1;
  float y1 = nx2 * z0 - nx0 * z2;
  float y2 = nx0 * z1 - nx1 * z0;
  float yi = rsqrtf(y0 * y0 + y1 * y1 + y2 * y2);
  y0 *= yi; y1 *= yi; y2 *= yi;
  float* o = out + (size_t)e * 9;
  o[0] = z0;  o[1] = z1;  o[2] = z2;
  o[3] = nx0; o[4] = nx1; o[5] = nx2;
  o[6] = -y0; o[7] = -y1; o[8] = -y2;
}

// ---------------------------------------------------------------------------
// 1. zero qb/kb region (so padded head-dim columns are exactly 0)
// ---------------------------------------------------------------------------
__global__ __launch_bounds__(256) void zero_kernel(uv4* __restrict__ p, int n4) {
  int i = blockIdx.x * 256 + threadIdx.x;
  if (i < n4) {
    uv4 z; z[0] = 0u; z[1] = 0u; z[2] = 0u; z[3] = 0u;
    p[i] = z;
  }
}

// ---------------------------------------------------------------------------
// 2. stage weights f32 -> bf16 : Wb[mode][l][o][c]
// ---------------------------------------------------------------------------
__global__ __launch_bounds__(256) void stage_w_kernel(const float* __restrict__ wq,
                                                      const float* __restrict__ wk,
                                                      const float* __restrict__ wv,
                                                      const float* __restrict__ wo,
                                                      unsigned short* __restrict__ wb) {
  int i = blockIdx.x * 256 + threadIdx.x;
  if (i >= 3 * 128 * 128) return;
  wb[i]                 = f2bf(wq[i]);
  wb[49152 + i]         = f2bf(wk[i]);
  wb[2 * 49152 + i]     = f2bf(wv[i]);
  wb[3 * 49152 + i]     = f2bf(wo[i]);
}

// ---------------------------------------------------------------------------
// 3. stage x f32 [N,D] -> xt bf16 per-l [N*d, C] (c contiguous)
// ---------------------------------------------------------------------------
__global__ __launch_bounds__(256) void stage_x_kernel(const float* __restrict__ x,
                                                      unsigned short* __restrict__ xt) {
  size_t i = (size_t)blockIdx.x * 256 + threadIdx.x;   // over N*D
  int n = (int)(i / DD), f = (int)(i % DD);
  xt[irrep_index(n, f)] = f2bf(x[i]);
}

// ---------------------------------------------------------------------------
// 4/6. irreps GEMM, templated on l-block (constant d) and destination mode:
//      per-l [rows,128] x [128,128]^T, WMMA bf16 -> f32
//      MODE: 0=Q, 1=K (write [b,h,m,160]), 2=V (write [b,h,160,m]),
//            3=O (write d_out f32 = x + attn_out)
// ---------------------------------------------------------------------------
template <int LI, int MODE>
__global__ __launch_bounds__(256) void gemm_kernel(const unsigned short* __restrict__ src,
                                                   const unsigned short* __restrict__ Wb,
                                                   unsigned short* __restrict__ dstQK,
                                                   float* __restrict__ dstO,
                                                   const float* __restrict__ xres) {
  __shared__ alignas(16) unsigned short As[16 * 128];

  constexpr int    d     = 2 * LI + 1;
  constexpr int    offf  = (LI == 0) ? 0 : ((LI == 1) ? 128 : 512);
  constexpr size_t lbase = (LI == 0) ? 0 : ((LI == 1) ? 1048576 : 4194304);

  const int    lt       = blockIdx.x;                 // local 16-row tile
  const size_t baseElem = lbase + (size_t)lt * 2048;  // 16 rows * 128 c

  // ---- stage A tile (16x128 bf16 = 4KB) into LDS via Tensor Data Mover ----
#if __has_builtin(__builtin_amdgcn_tensor_load_to_lds)
  if (threadIdx.x < 32) {
    unsigned long long ga = (unsigned long long)(uintptr_t)(src + baseElem);
    unsigned lds0 = (unsigned)(uintptr_t)&As[0];
    u32x4 g0;
    g0[0] = 1u;                                      // count=1 (valid D#)
    g0[1] = lds0;                                    // lds_addr
    g0[2] = (unsigned)(ga & 0xffffffffu);            // global_addr lo
    g0[3] = (unsigned)((ga >> 32) & 0x01ffffffu) | (2u << 30);  // addr hi | type=2
    i32x8 g1;
    g1[0] = (1 << 16);                               // data_size = 2 bytes
    g1[1] = (128 << 16);                             // tensor_dim0 = 128
    g1[2] = (16 << 16);                              // tensor_dim1 = 16
    g1[3] = (128 << 16);                             // tile_dim0   = 128
    g1[4] = 16;                                      // tile_dim1=16, tile_dim2=0
    g1[5] = 128;                                     // tensor_dim0_stride = 128
    g1[6] = 0;
    g1[7] = 0;
    i32x4 gz; gz[0] = 0; gz[1] = 0; gz[2] = 0; gz[3] = 0;
#if __has_include(<hip/amd_detail/amd_gfx1250_TDM.h>)
    i32x8 gz8;
#pragma unroll
    for (int i = 0; i < 8; ++i) gz8[i] = 0;
    __builtin_amdgcn_tensor_load_to_lds(g0, g1, gz, gz, gz8, 0);
#else
    __builtin_amdgcn_tensor_load_to_lds(g0, g1, gz, gz, 0);
#endif
    __builtin_amdgcn_s_wait_tensorcnt(0);
  }
#else
  {
    const uv4* s = reinterpret_cast<const uv4*>(src + baseElem);
    uv4* dp = reinterpret_cast<uv4*>(&As[0]);
    dp[threadIdx.x] = s[threadIdx.x];                // 256 * 16B = 4KB
  }
#endif
  __syncthreads();

  const int w    = threadIdx.x >> 5;                 // 8 waves: out cols w*16..
  const int lane = threadIdx.x & 31;
  const int ln   = lane & 15;
  const int hihf = lane >> 4;
  const int klo  = hihf ? 16 : 0;                    // A-frag K offset
  const int kofB = hihf ? 16 : 0;                    // B-frag K offset

  const unsigned short* Wl = Wb + (size_t)LI * 16384;
  v8f acc = vzero8();
#pragma unroll
  for (int kc = 0; kc < 4; ++kc) {
    v16bf a = load_frag2(&As[ln * 128 + kc * 32 + klo],
                         &As[ln * 128 + kc * 32 + klo + 8]);
    v16bf bf = load_frag32B(Wl + (size_t)(w * 16 + ln) * 128 + kc * 32 + kofB);
    acc = wmma_bf16(a, bf, acc);
  }

  const float sc = 0.08838834764831845f;             // 1/sqrt(128)
  const int oc = w * 16 + ln;
#pragma unroll
  for (int r = 0; r < 8; ++r) {
    int grow = lt * 16 + r + 8 * hihf;               // region-local row
    int n = grow / d, m = grow - n * d;              // d is constexpr
    int f = offf + oc * d + m;
    float val = acc[r] * sc;
    if (MODE == 3) {
      size_t io = (size_t)n * DD + f;
      dstO[io] = xres[io] + val;
    } else {
      int b  = n >> 9, mr = n & 511;
      int h  = f / HD, dw = f - h * HD;
      size_t io;
      if (MODE == 2)
        io = ((size_t)(b * HH + h) * HDP + dw) * MM + mr;     // V transposed
      else
        io = ((size_t)(b * HH + h) * MM + mr) * HDP + dw;     // Q / K
      dstQK[io] = f2bf(val);
    }
  }
}

// ---------------------------------------------------------------------------
// 5. flash attention: one wave per (b, h, 16-q tile); writes ot (per-l bf16)
// ---------------------------------------------------------------------------
__global__ __launch_bounds__(32) void attn_kernel(const unsigned short* __restrict__ qb,
                                                  const unsigned short* __restrict__ kb,
                                                  const unsigned short* __restrict__ vt,
                                                  unsigned short* __restrict__ ot) {
  __shared__ alignas(16) unsigned short sP[16 * 32];

  const int lane = threadIdx.x & 31;
  const int ln   = lane & 15;
  const int hihf = lane >> 4;
  const int blk  = blockIdx.x;                       // 0..4095
  const int qt   = blk & 31;
  const int h    = (blk >> 5) & 7;
  const int b    = blk >> 8;

  const size_t bh = (size_t)(b * HH + h);
  const unsigned short* qbase = qb + bh * MM * HDP;
  const unsigned short* kbase = kb + bh * MM * HDP;
  const unsigned short* vbase = vt + bh * HDP * MM;

  // resident Q fragments: 5 K-chunks over padded head dim
  v16bf Qf[5];
  {
    const unsigned short* qp = qbase + (size_t)(qt * 16 + ln) * HDP;
    const int klo = hihf ? 16 : 0;
#pragma unroll
    for (int kc = 0; kc < 5; ++kc)
      Qf[kc] = load_frag2(qp + kc * 32 + klo, qp + kc * 32 + klo + 8);
  }

  v8f O[9];
#pragma unroll
  for (int t = 0; t < 9; ++t) O[t] = vzero8();
  float mrow[8], lrow[8];
#pragma unroll
  for (int r = 0; r < 8; ++r) { mrow[r] = -1e30f; lrow[r] = 0.0f; }

  const float scale = 0.08333333333333333f;          // 1/sqrt(144)
  const int koff = hihf ? 16 : 0;

  for (int kb2 = 0; kb2 < 16; ++kb2) {               // 16 blocks of 32 keys
    if (kb2 + 1 < 16)
      __builtin_prefetch(kbase + (size_t)((kb2 + 1) * 32 + ln) * HDP, 0, 1);

    // S = Q K^T for two 16-key halves
    v8f S0 = vzero8(), S1 = vzero8();
#pragma unroll
    for (int kc = 0; kc < 5; ++kc) {
      v16bf K0 = load_frag32B(kbase + (size_t)(kb2 * 32 + ln) * HDP + kc * 32 + koff);
      v16bf K1 = load_frag32B(kbase + (size_t)(kb2 * 32 + 16 + ln) * HDP + kc * 32 + koff);
      S0 = wmma_bf16(Qf[kc], K0, S0);
      S1 = wmma_bf16(Qf[kc], K1, S1);
    }

    // streaming softmax update (row stats replicated per 16-lane half)
    float alpha[8];
#pragma unroll
    for (int r = 0; r < 8; ++r) {
      float a0 = S0[r] * scale, a1 = S1[r] * scale;
      float v = fmaxf(a0, a1);
      v = fmaxf(v, __shfl_xor(v, 1));
      v = fmaxf(v, __shfl_xor(v, 2));
      v = fmaxf(v, __shfl_xor(v, 4));
      v = fmaxf(v, __shfl_xor(v, 8));
      float mn = fmaxf(mrow[r], v);
      alpha[r] = __expf(mrow[r] - mn);
      mrow[r] = mn;
      float p0 = __expf(a0 - mn), p1 = __expf(a1 - mn);
      S0[r] = p0; S1[r] = p1;
      float rs = p0 + p1;
      rs += __shfl_xor(rs, 1);
      rs += __shfl_xor(rs, 2);
      rs += __shfl_xor(rs, 4);
      rs += __shfl_xor(rs, 8);
      lrow[r] = lrow[r] * alpha[r] + rs;
    }
#pragma unroll
    for (int t = 0; t < 9; ++t)
#pragma unroll
      for (int r = 0; r < 8; ++r) O[t][r] *= alpha[r];

    // P (D-layout f32) -> LDS bf16 [16 q][32 key] -> A fragment
#pragma unroll
    for (int r = 0; r < 8; ++r) {
      int row = r + 8 * hihf;
      sP[row * 32 + ln]      = f2bf(S0[r]);
      sP[row * 32 + 16 + ln] = f2bf(S1[r]);
    }
    asm volatile("s_wait_dscnt 0" ::: "memory");
    v16bf Pf;
    {
      const int klo = hihf ? 16 : 0;
      const unsigned short* pp = &sP[ln * 32 + klo];
      Pf = load_frag2(pp, pp + 8);
    }

    // O += P * V  (V pre-transposed: contiguous in key)
#pragma unroll
    for (int t = 0; t < 9; ++t) {
      v16bf Vf = load_frag32B(vbase + (size_t)(t * 16 + ln) * MM + kb2 * 32 + koff);
      O[t] = wmma_bf16(Pf, Vf, O[t]);
    }
  }

  // normalize and scatter into per-l layout for the W_o GEMM
  const int nbase = b * MM + qt * 16;
#pragma unroll
  for (int t = 0; t < 9; ++t) {
#pragma unroll
    for (int r = 0; r < 8; ++r) {
      int nn = nbase + r + 8 * hihf;
      float val = O[t][r] / lrow[r];
      int f = h * HD + t * 16 + ln;
      ot[irrep_index(nn, f)] = f2bf(val);
    }
  }
}

// ---------------------------------------------------------------------------
// 7. LayerNorm + SiLU on first 128 channels, in place on d_out (one wave/row)
// ---------------------------------------------------------------------------
__global__ __launch_bounds__(256) void ln_kernel(float* __restrict__ y,
                                                 const float* __restrict__ g,
                                                 const float* __restrict__ bta) {
  int w = threadIdx.x >> 5, lane = threadIdx.x & 31;
  int n = blockIdx.x * 8 + w;
  float* row = y + (size_t)n * DD;
  float v0[4];
  float s = 0.0f, s2 = 0.0f;
#pragma unroll
  for (int i = 0; i < 4; ++i) {
    float x = row[lane * 4 + i];
    v0[i] = x; s += x; s2 += x * x;
  }
#pragma unroll
  for (int off = 1; off < 32; off <<= 1) {
    s  += __shfl_xor(s, off);
    s2 += __shfl_xor(s2, off);
  }
  float mu  = s * (1.0f / 128.0f);
  float var = s2 * (1.0f / 128.0f) - mu * mu;
  float inv = rsqrtf(var + 1e-5f);
#pragma unroll
  for (int i = 0; i < 4; ++i) {
    int c = lane * 4 + i;
    float z = (v0[i] - mu) * inv * g[c] + bta[c];
    row[c] = z / (1.0f + __expf(-z));
  }
}

// ---------------------------------------------------------------------------
template <int MODE>
static void launch_gemm(const unsigned short* src, const unsigned short* Wb,
                        unsigned short* dstQK, float* dstO, const float* xres,
                        hipStream_t stream) {
  gemm_kernel<0, MODE><<<512,  256, 0, stream>>>(src, Wb, dstQK, dstO, xres);
  gemm_kernel<1, MODE><<<1536, 256, 0, stream>>>(src, Wb, dstQK, dstO, xres);
  gemm_kernel<2, MODE><<<2560, 256, 0, stream>>>(src, Wb, dstQK, dstO, xres);
}

extern "C" void kernel_launch(void* const* d_in, const int* in_sizes, int n_in,
                              void* d_out, int out_size, void* d_ws, size_t ws_size,
                              hipStream_t stream) {
  (void)in_sizes; (void)n_in; (void)out_size; (void)ws_size;

  const float* x     = (const float*)d_in[0];
  const float* ev    = (const float*)d_in[1];
  const float* er    = (const float*)d_in[2];
  const float* wq    = (const float*)d_in[3];
  const float* wk    = (const float*)d_in[4];
  const float* wv    = (const float*)d_in[5];
  const float* wo    = (const float*)d_in[6];
  const float* gamma = (const float*)d_in[7];
  const float* beta  = (const float*)d_in[8];

  float* out = (float*)d_out;
  char*  ws  = (char*)d_ws;

  const size_t OFF_XT = 0;
  const size_t OFF_QB = 18874368;
  const size_t OFF_KB = OFF_QB + 20971520;
  const size_t OFF_VT = OFF_KB + 20971520;
  const size_t OFF_OT = OFF_VT + 20971520;
  const size_t OFF_WB = OFF_OT + 18874368;

  unsigned short* xt = (unsigned short*)(ws + OFF_XT);
  unsigned short* qb = (unsigned short*)(ws + OFF_QB);
  unsigned short* kbp= (unsigned short*)(ws + OFF_KB);
  unsigned short* vt = (unsigned short*)(ws + OFF_VT);
  unsigned short* ot = (unsigned short*)(ws + OFF_OT);
  unsigned short* wb = (unsigned short*)(ws + OFF_WB);

  // 0. rot (independent)
  rot_kernel<<<EE / 256, 256, 0, stream>>>(ev, er, out + (size_t)NN * DD);

  // 1. zero qb + kb (contiguous 40MB region; padding columns must be 0)
  zero_kernel<<<10240, 256, 0, stream>>>((uv4*)(ws + OFF_QB), 2621440);

  // 2. weights -> bf16
  stage_w_kernel<<<192, 256, 0, stream>>>(wq, wk, wv, wo, wb);

  // 3. x -> xt
  stage_x_kernel<<<36864, 256, 0, stream>>>(x, xt);

  // 4. projections (Q, K, V)
  launch_gemm<0>(xt, wb,             qb,  nullptr, nullptr, stream);
  launch_gemm<1>(xt, wb + 49152,     kbp, nullptr, nullptr, stream);
  launch_gemm<2>(xt, wb + 2 * 49152, vt,  nullptr, nullptr, stream);

  // 5. attention
  attn_kernel<<<4096, 32, 0, stream>>>(qb, kbp, vt, ot);

  // 6. output projection + residual -> d_out (f32)
  launch_gemm<3>(ot, wb + 3 * 49152, nullptr, out, x, stream);

  // 7. LN + SiLU on first 128 channels
  ln_kernel<<<1024, 256, 0, stream>>>(out, gamma, beta);
}
